// PoseBlockSpatial_59304908423913
// MI455X (gfx1250) — matrix operand order repaired and checked by reading, not compiled, not run on hardware
//
#include <hip/hip_runtime.h>
#include <math.h>
#include <stdint.h>

// ---------------------------------------------------------------------------
// CDNA5 (gfx1250) WMMA types
// ---------------------------------------------------------------------------
typedef __attribute__((ext_vector_type(16))) __bf16 v16bf;
typedef __attribute__((ext_vector_type(8)))  float  v8f;
typedef __attribute__((ext_vector_type(8)))  __bf16 bfx8;

#define D_MODEL 768
#define D_HID   3072
#define NB      8
#define NT      8
#define NTOK    197               // 14*14 + 1
#define BTB     64                // B*T
#define NSEQ    1569              // T*H*W + 1
#define M_BT    (BTB * NTOK)      // 12608 rows in frame-token space
#define M_Y     (NB * NSEQ)       // 12552 rows in original-token space
#define NHEADS  12
#define HD      64

// ---------------------------------------------------------------------------
// CDNA5 async global->LDS copy (ASYNCcnt-tracked), with plain-load fallback.
// Builtin signature (from hipcc diagnostic): param0 = int4 addrspace(1)*,
// param1 = int4 addrspace(3)*, then imm offset, imm cpol.
// ---------------------------------------------------------------------------
#if __has_builtin(__builtin_amdgcn_global_load_async_to_lds_b128) && \
    __has_builtin(__builtin_amdgcn_s_wait_asynccnt)
#define HAVE_ASYNC_LDS 1
#endif

typedef int v4i __attribute__((vector_size(16)));
typedef __attribute__((address_space(1))) v4i gv4i_t;
typedef __attribute__((address_space(3))) v4i lv4i_t;

__device__ __forceinline__ void cp16_g2l(const __bf16* g, __bf16* l) {
#ifdef HAVE_ASYNC_LDS
  // Generic LDS addresses carry the LDS byte offset in the low 32 bits;
  // AS3 pointers are 32-bit. Integer casts keep these plain reinterprets.
  __builtin_amdgcn_global_load_async_to_lds_b128(
      (gv4i_t*)(uintptr_t)g, (lv4i_t*)(unsigned)(uintptr_t)l, 0, 0);
#else
  *(bfx8*)l = *(const bfx8*)g;
#endif
}
__device__ __forceinline__ void wait_async_le4() {
#ifdef HAVE_ASYNC_LDS
  __builtin_amdgcn_s_wait_asynccnt(4);
#endif
}
__device__ __forceinline__ void wait_async_le0() {
#ifdef HAVE_ASYNC_LDS
  __builtin_amdgcn_s_wait_asynccnt(0);
#endif
}

__device__ __forceinline__ float epi_apply(int EPI, float v, int row, int col,
                                           void* OutV, long oidx) {
  if (EPI == 0) {
    ((__bf16*)OutV)[oidx] = (__bf16)v;
  } else if (EPI == 1) {
    ((float*)OutV)[oidx] = v;
  } else if (EPI == 2) {
    if (row == 0 && col == 0) v = 100.0f;          // cls->cls gate forced open
    ((float*)OutV)[oidx] = 1.0f / (1.0f + expf(-v));
  } else if (EPI == 3) {
    const float g = 0.5f * v * (1.0f + erff(v * 0.70710678118654752f));
    ((__bf16*)OutV)[oidx] = (__bf16)g;
  } else {
    float* O = (float*)OutV;
    O[oidx] = O[oidx] + v;                         // add MLP branch into y
  }
  return v;
}

// ---------------------------------------------------------------------------
// Big-tile WMMA GEMM: Out = epi( A[M,K]bf16 @ W[N,K]^T bf16 + bias )
// 256 threads (8 waves), tile 128x128, K-step 32, double-buffered async LDS.
// Requires: K % 32 == 0, N % 128 == 0 (M may be ragged).
// Each wave owns a 32x64 strip: 2 A-frags x 4 B-frags -> 8 WMMA / K-step.
// ---------------------------------------------------------------------------
template <int EPI>
__global__ __launch_bounds__(256)
void wmma_gemm_128(const __bf16* __restrict__ A, const __bf16* __restrict__ W,
                   const float* __restrict__ bias, void* __restrict__ OutV,
                   int M, int N, int K) {
  __shared__ __bf16 Abuf[2][128][40];   // 40-elem row stride: 16B-aligned, bank-spread
  __shared__ __bf16 Bbuf[2][128][40];

  const int tid  = threadIdx.x;
  const int lane = tid & 31;
  const int wave = tid >> 5;
  const int lrow = lane & 15;
  const int lhi  = lane >> 4;
  const int wm   = (wave >> 1) * 32;    // wave's row offset within tile
  const int wn   = (wave & 1) * 64;     // wave's col offset within tile
  const int rowBase = blockIdx.y * 128;
  const int colBase = blockIdx.x * 128;

  // staging: each thread copies 16 bf16 (2 x b128) per matrix per K-step
  const int sr = tid >> 1;              // 0..127 (tile row)
  const int sh = (tid & 1) * 16;        // element offset 0 or 16
  int ar = rowBase + sr; if (ar > M - 1) ar = M - 1;   // clamp ragged M
  const long aoff = (long)ar * K + sh;
  const long boff = (long)(colBase + sr) * K + sh;

  auto stage = [&](int buf, int k0) {
    cp16_g2l(A + aoff + k0,     &Abuf[buf][sr][sh]);
    cp16_g2l(A + aoff + k0 + 8, &Abuf[buf][sr][sh + 8]);
    cp16_g2l(W + boff + k0,     &Bbuf[buf][sr][sh]);
    cp16_g2l(W + boff + k0 + 8, &Bbuf[buf][sr][sh + 8]);
  };

  v8f acc[2][4] = {};

  const int nk = K >> 5;
  stage(0, 0);
  for (int ks = 0; ks < nk; ks++) {
    const int buf = ks & 1;
    if (ks + 1 < nk) { stage(buf ^ 1, (ks + 1) << 5); wait_async_le4(); }
    else             { wait_async_le0(); }
    __syncthreads();

    v16bf afr[2];
    #pragma unroll
    for (int m = 0; m < 2; m++) {
      const __bf16* ap = &Abuf[buf][wm + m * 16 + lrow][lhi * 16];
      #pragma unroll
      for (int i = 0; i < 8; i++) { afr[m][i] = ap[i]; afr[m][8 + i] = ap[8 + i]; }
    }
    #pragma unroll
    for (int t = 0; t < 4; t++) {
      v16bf b;
      const __bf16* bp = &Bbuf[buf][wn + t * 16 + lrow][lhi * 16];
      #pragma unroll
      for (int i = 0; i < 16; i++) b[i] = bp[i];
      #pragma unroll
      for (int m = 0; m < 2; m++)
        acc[m][t] = __builtin_amdgcn_wmma_f32_16x16x32_bf16(
            false, afr[m], false, b, (short)0, acc[m][t], false, false);
    }
    __syncthreads();
  }

  // Epilogue. C/D layout: VGPR i -> M = lhi*8 + i, N = lrow (per 16x16 tile).
  #pragma unroll
  for (int m = 0; m < 2; m++) {
    const int row0 = rowBase + wm + m * 16 + lhi * 8;
    #pragma unroll
    for (int t = 0; t < 4; t++) {
      const int col = colBase + wn + t * 16 + lrow;
      if (col >= N) continue;
      const float bv = bias ? bias[col] : 0.0f;
      #pragma unroll
      for (int i = 0; i < 8; i++) {
        const int row = row0 + i;
        if (row >= M) continue;
        epi_apply(EPI, acc[m][t][i] + bv, row, col, OutV, (long)row * N + col);
      }
    }
  }
}

// ---------------------------------------------------------------------------
// 64x64-tile WMMA GEMM (batched, ragged M/N/K) -- used for the 197x197 lm GEMM.
// ---------------------------------------------------------------------------
template <int EPI>
__global__ __launch_bounds__(128)
void wmma_gemm_64(const __bf16* __restrict__ A, const __bf16* __restrict__ W,
                  const float* __restrict__ bias, void* __restrict__ OutV,
                  int M, int N, int K,
                  long batchStrideA, long batchStrideW, long batchStrideO) {
  __shared__ __bf16 Atile[64][32];
  __shared__ __bf16 Btile[64][32];

  const int tid  = threadIdx.x;
  const int lane = tid & 31;
  const int wave = tid >> 5;
  const int lrow = lane & 15;
  const int lhi  = lane >> 4;

  const int bz = blockIdx.z;
  A += (long)bz * batchStrideA;
  W += (long)bz * batchStrideW;
  const int rowBase = blockIdx.y * 64;
  const int colBase = blockIdx.x * 64;

  v8f acc[4] = {};

  for (int k0 = 0; k0 < K; k0 += 32) {
    if (k0 + 32 < K) {
      __builtin_prefetch(&A[(long)(rowBase + (tid >> 1)) * K + k0 + 32], 0, 1);
      __builtin_prefetch(&W[(long)(colBase + (tid >> 1)) * K + k0 + 32], 0, 1);
    }
    for (int i = tid; i < 64 * 32; i += 128) {
      const int r = i >> 5, c = i & 31;
      const int gr = rowBase + r, gc = k0 + c;
      Atile[r][c] = (gr < M && gc < K) ? A[(long)gr * K + gc] : (__bf16)0.0f;
      const int wr = colBase + r;
      Btile[r][c] = (wr < N && gc < K) ? W[(long)wr * K + gc] : (__bf16)0.0f;
    }
    __syncthreads();

    v16bf a;
    {
      const __bf16* ap = &Atile[wave * 16 + lrow][lhi * 16];
      #pragma unroll
      for (int i = 0; i < 8; i++) { a[i] = ap[i]; a[8 + i] = ap[8 + i]; }
    }
    #pragma unroll
    for (int t = 0; t < 4; t++) {
      v16bf b;
      const __bf16* bp = &Btile[t * 16 + lrow][lhi * 16];
      #pragma unroll
      for (int i = 0; i < 16; i++) b[i] = bp[i];
      acc[t] = __builtin_amdgcn_wmma_f32_16x16x32_bf16(
          false, a, false, b, (short)0, acc[t], false, false);
    }
    __syncthreads();
  }

  const int row0 = rowBase + wave * 16 + lhi * 8;
  #pragma unroll
  for (int t = 0; t < 4; t++) {
    const int col = colBase + t * 16 + lrow;
    if (col >= N) continue;
    const float bv = bias ? bias[col] : 0.0f;
    #pragma unroll
    for (int i = 0; i < 8; i++) {
      const int row = row0 + i;
      if (row >= M) continue;
      epi_apply(EPI, acc[t][i] + bv, row, col, OutV,
                (long)bz * batchStrideO + (long)row * N + col);
    }
  }
}

// ---------------------------------------------------------------------------
// Fused gated attention: one wave per (bt, head, 16-query tile).
// scores = q @ k^T * hd^-0.5 ; p = softmax(scores) * lm ; o = p @ v
// ---------------------------------------------------------------------------
__global__ __launch_bounds__(32)
void attn_fused(const __bf16* __restrict__ qkv, const float* __restrict__ lm,
                __bf16* __restrict__ o) {
  const int qt = blockIdx.x;    // 0..12
  const int h  = blockIdx.y;    // 0..11
  const int bt = blockIdx.z;    // 0..63
  const int lane = threadIdx.x;
  const int lrow = lane & 15;
  const int lhi  = lane >> 4;
  const int q0 = qt * 16;

  __shared__ __bf16 qT[16][64];
  __shared__ __bf16 kT[16][64];
  __shared__ float  sc[16][212];
  __shared__ __bf16 sbf[16][224];
  __shared__ __bf16 vT[64][32];   // transposed: [hd][key]

  const long rowStride = 3 * D_MODEL;  // qkv row = [3][12][64]
  const __bf16* qbase = qkv + (long)bt * NTOK * rowStride + h * HD;
  const __bf16* kbase = qbase + D_MODEL;
  const __bf16* vbase = qbase + 2 * D_MODEL;

  for (int i = lane; i < 16 * 64; i += 32) {
    const int r = i >> 6, c = i & 63;
    const int q = q0 + r;
    qT[r][c] = (q < NTOK) ? qbase[(long)q * rowStride + c] : (__bf16)0.0f;
  }
  __syncthreads();

  for (int kt = 0; kt < 13; kt++) {
    for (int i = lane; i < 16 * 64; i += 32) {
      const int r = i >> 6, c = i & 63;
      const int kk = kt * 16 + r;
      kT[r][c] = (kk < NTOK) ? kbase[(long)kk * rowStride + c] : (__bf16)0.0f;
    }
    __syncthreads();
    v8f accS = {};
    #pragma unroll
    for (int s = 0; s < 2; s++) {
      v16bf a, b;
      const __bf16* ap = &qT[lrow][s * 32 + lhi * 16];
      #pragma unroll
      for (int i = 0; i < 8; i++) { a[i] = ap[i]; a[8 + i] = ap[8 + i]; }
      const __bf16* bp = &kT[lrow][s * 32 + lhi * 16];
      #pragma unroll
      for (int i = 0; i < 16; i++) b[i] = bp[i];
      accS = __builtin_amdgcn_wmma_f32_16x16x32_bf16(
          false, a, false, b, (short)0, accS, false, false);
    }
    #pragma unroll
    for (int i = 0; i < 8; i++)
      sc[lhi * 8 + i][kt * 16 + lrow] = accS[i] * 0.125f;   // hd^-0.5
    __syncthreads();
  }

  for (int i = lane; i < 16 * 224; i += 32) {
    const int r = i / 224, c = i % 224;
    sbf[r][c] = (__bf16)0.0f;
  }
  __syncthreads();
  if (lane < 16 && (q0 + lane) < NTOK) {
    const int r = lane;
    float m = -1e30f;
    for (int c = 0; c < NTOK; c++) m = fmaxf(m, sc[r][c]);
    float s = 0.0f;
    for (int c = 0; c < NTOK; c++) s += expf(sc[r][c] - m);
    const float inv = 1.0f / s;
    const float* lmrow = lm + ((long)bt * NTOK + q0 + r) * NTOK;
    for (int c = 0; c < NTOK; c++)
      sbf[r][c] = (__bf16)(expf(sc[r][c] - m) * inv * lmrow[c]);
  }
  __syncthreads();

  v8f accO[4] = {};
  for (int kt = 0; kt < 7; kt++) {
    for (int i = lane; i < 32 * 64; i += 32) {
      const int kk = i >> 6, c = i & 63;
      const int g = kt * 32 + kk;
      vT[c][kk] = (g < NTOK) ? vbase[(long)g * rowStride + c] : (__bf16)0.0f;
    }
    __syncthreads();
    v16bf a;
    const __bf16* ap = &sbf[lrow][kt * 32 + lhi * 16];
    #pragma unroll
    for (int i = 0; i < 8; i++) { a[i] = ap[i]; a[8 + i] = ap[8 + i]; }
    #pragma unroll
    for (int t = 0; t < 4; t++) {
      v16bf b;
      const __bf16* bp = &vT[t * 16 + lrow][lhi * 16];
      #pragma unroll
      for (int i = 0; i < 16; i++) b[i] = bp[i];
      accO[t] = __builtin_amdgcn_wmma_f32_16x16x32_bf16(
          false, a, false, b, (short)0, accO[t], false, false);
    }
    __syncthreads();
  }

  #pragma unroll
  for (int t = 0; t < 4; t++) {
    #pragma unroll
    for (int i = 0; i < 8; i++) {
      const int q = q0 + lhi * 8 + i;
      if (q < NTOK)
        o[((long)bt * NTOK + q) * D_MODEL + h * HD + t * 16 + lrow] =
            (__bf16)accO[t][i];
    }
  }
}

// ---------------------------------------------------------------------------
// LayerNorm (population var, eps 1e-5): f32 in -> bf16 out. One block / token.
// ---------------------------------------------------------------------------
__global__ __launch_bounds__(256)
void ln_to_bf16(const float* __restrict__ in, const float* __restrict__ g,
                const float* __restrict__ b, __bf16* __restrict__ out) {
  const int tok = blockIdx.x;
  const int tid = threadIdx.x;
  __shared__ float r1[256], r2[256];
  const float* row = in + (long)tok * D_MODEL;
  float s = 0.f, s2 = 0.f;
  for (int i = tid; i < D_MODEL; i += 256) { float v = row[i]; s += v; s2 += v * v; }
  r1[tid] = s; r2[tid] = s2;
  __syncthreads();
  for (int off = 128; off > 0; off >>= 1) {
    if (tid < off) { r1[tid] += r1[tid + off]; r2[tid] += r2[tid + off]; }
    __syncthreads();
  }
  const float mean = r1[0] * (1.0f / D_MODEL);
  const float var  = r2[0] * (1.0f / D_MODEL) - mean * mean;
  const float rstd = rsqrtf(var + 1e-5f);
  __bf16* orow = out + (long)tok * D_MODEL;
  for (int i = tid; i < D_MODEL; i += 256)
    orow[i] = (__bf16)((row[i] - mean) * rstd * g[i] + b[i]);
}

// ---------------------------------------------------------------------------
// Token regather: x(B, 1+HW*T, D) -> xt(B*T, 197, D)  (f32 + bf16 copies)
// ---------------------------------------------------------------------------
__global__ void build_xt(const float* __restrict__ x, float* __restrict__ xtf,
                         __bf16* __restrict__ xtb) {
  const long idx = (long)blockIdx.x * 256 + threadIdx.x;
  if (idx >= (long)M_BT * D_MODEL) return;
  const int d = (int)(idx % D_MODEL);
  const long r = idx / D_MODEL;
  const int n  = (int)(r % NTOK);
  const int bt = (int)(r / NTOK);
  const int b = bt >> 3, t = bt & 7;
  long src;
  if (n == 0) src = ((long)b * NSEQ) * D_MODEL + d;
  else        src = ((long)b * NSEQ + 1 + (long)(n - 1) * NT + t) * D_MODEL + d;
  const float v = x[src];
  xtf[idx] = v;
  xtb[idx] = (__bf16)v;
}

// ---------------------------------------------------------------------------
// y = x + scatter(proj): cls = mean over frames; spatial re-interleaved.
// ---------------------------------------------------------------------------
__global__ void scatter_residual(const float* __restrict__ x,
                                 const float* __restrict__ proj,
                                 float* __restrict__ y) {
  const long idx = (long)blockIdx.x * 256 + threadIdx.x;
  if (idx >= (long)M_Y * D_MODEL) return;
  const int d = (int)(idx % D_MODEL);
  const long r = idx / D_MODEL;
  const int tok = (int)(r % NSEQ);
  const int b   = (int)(r / NSEQ);
  float v;
  if (tok == 0) {
    float s = 0.f;
    for (int t = 0; t < NT; t++)
      s += proj[((long)(b * NT + t) * NTOK) * D_MODEL + d];
    v = s * (1.0f / NT);
  } else {
    const int sidx = tok - 1;
    const int hw = sidx >> 3, t = sidx & 7;
    v = proj[((long)(b * NT + t) * NTOK + 1 + hw) * D_MODEL + d];
  }
  y[idx] = x[idx] + v;
}

__global__ void f32_to_bf16(const float* __restrict__ in,
                            __bf16* __restrict__ out, long n) {
  const long i = (long)blockIdx.x * 256 + threadIdx.x;
  if (i < n) out[i] = (__bf16)in[i];
}

// ---------------------------------------------------------------------------
// Host launcher
// ---------------------------------------------------------------------------
extern "C" void kernel_launch(void* const* d_in, const int* in_sizes, int n_in,
                              void* d_out, int out_size, void* d_ws, size_t ws_size,
                              hipStream_t stream) {
  (void)in_sizes; (void)n_in; (void)out_size; (void)ws_size;

  const float* x      = (const float*)d_in[0];
  const float* W_mq   = (const float*)d_in[1];
  const float* b_mq   = (const float*)d_in[2];
  const float* W_mk   = (const float*)d_in[3];
  const float* b_mk   = (const float*)d_in[4];
  const float* g1     = (const float*)d_in[5];
  const float* be1    = (const float*)d_in[6];
  const float* W_qkv  = (const float*)d_in[7];
  const float* W_proj = (const float*)d_in[8];
  const float* b_proj = (const float*)d_in[9];
  const float* g2     = (const float*)d_in[10];
  const float* be2    = (const float*)d_in[11];
  const float* W_fc1  = (const float*)d_in[12];
  const float* b_fc1  = (const float*)d_in[13];
  const float* W_fc2  = (const float*)d_in[14];
  const float* b_fc2  = (const float*)d_in[15];

  float* yOut  = (float*)d_out;                         // (8,1569,768)
  float* lmOut = yOut + (size_t)M_Y * D_MODEL;          // (64,197,197)

  char*  ws  = (char*)d_ws;
  size_t off = 0;
  auto alloc = [&](size_t bytes) -> void* {
    void* p = ws + off;
    off += (bytes + 255) & ~(size_t)255;
    return p;
  };
  float*  xtf   = (float*) alloc((size_t)M_BT * D_MODEL * 4);
  __bf16* xtb   = (__bf16*)alloc((size_t)M_BT * D_MODEL * 2);
  __bf16* qmb   = (__bf16*)alloc((size_t)M_BT * D_MODEL * 2);
  __bf16* kmb   = (__bf16*)alloc((size_t)M_BT * D_MODEL * 2);
  __bf16* xnb   = (__bf16*)alloc((size_t)M_BT * D_MODEL * 2);
  __bf16* qkvb  = (__bf16*)alloc((size_t)M_BT * 3 * D_MODEL * 2);
  __bf16* ob    = (__bf16*)alloc((size_t)M_BT * D_MODEL * 2);
  float*  projf = (float*) alloc((size_t)M_BT * D_MODEL * 4);
  __bf16* h1b   = (__bf16*)alloc((size_t)M_Y * D_MODEL * 2);
  __bf16* h2b   = (__bf16*)alloc((size_t)M_Y * D_HID * 2);
  __bf16* wmqb  = (__bf16*)alloc((size_t)D_MODEL * D_MODEL * 2);
  __bf16* wmkb  = (__bf16*)alloc((size_t)D_MODEL * D_MODEL * 2);
  __bf16* wqkvb = (__bf16*)alloc((size_t)3 * D_MODEL * D_MODEL * 2);
  __bf16* wprjb = (__bf16*)alloc((size_t)D_MODEL * D_MODEL * 2);
  __bf16* wfc1b = (__bf16*)alloc((size_t)D_HID * D_MODEL * 2);
  __bf16* wfc2b = (__bf16*)alloc((size_t)D_MODEL * D_HID * 2);

  auto cvt = [&](const float* src, __bf16* dst, long n) {
    f32_to_bf16<<<(int)((n + 255) / 256), 256, 0, stream>>>(src, dst, n);
  };
  cvt(W_mq,   wmqb,  (long)D_MODEL * D_MODEL);
  cvt(W_mk,   wmkb,  (long)D_MODEL * D_MODEL);
  cvt(W_qkv,  wqkvb, (long)3 * D_MODEL * D_MODEL);
  cvt(W_proj, wprjb, (long)D_MODEL * D_MODEL);
  cvt(W_fc1,  wfc1b, (long)D_HID * D_MODEL);
  cvt(W_fc2,  wfc2b, (long)D_MODEL * D_HID);

  build_xt<<<(int)(((long)M_BT * D_MODEL + 255) / 256), 256, 0, stream>>>(x, xtf, xtb);

  const dim3 blk128(128), blk256(256);
  auto grid128 = [](int M, int N) { return dim3(N / 128, (M + 127) / 128, 1); };

  // qm = xt @ W_mq^T + b_mq ; km = xt @ W_mk^T + b_mk   (bf16 out)
  wmma_gemm_128<0><<<grid128(M_BT, D_MODEL), blk256, 0, stream>>>(
      xtb, wmqb, b_mq, qmb, M_BT, D_MODEL, D_MODEL);
  wmma_gemm_128<0><<<grid128(M_BT, D_MODEL), blk256, 0, stream>>>(
      xtb, wmkb, b_mk, kmb, M_BT, D_MODEL, D_MODEL);

  // lm = sigmoid(qm @ km^T) batched over 64 frames; [b,0,0] forced to 100
  wmma_gemm_64<2><<<dim3(4, 4, BTB), blk128, 0, stream>>>(
      qmb, kmb, nullptr, lmOut, NTOK, NTOK, D_MODEL,
      (long)NTOK * D_MODEL, (long)NTOK * D_MODEL, (long)NTOK * NTOK);

  // xn = LN(xt)
  ln_to_bf16<<<M_BT, blk256, 0, stream>>>(xtf, g1, be1, xnb);

  // qkv = xn @ W_qkv^T  (no bias)
  wmma_gemm_128<0><<<grid128(M_BT, 3 * D_MODEL), blk256, 0, stream>>>(
      xnb, wqkvb, nullptr, qkvb, M_BT, 3 * D_MODEL, D_MODEL);

  // fused gated attention
  attn_fused<<<dim3(13, NHEADS, BTB), 32, 0, stream>>>(qkvb, lmOut, ob);

  // proj
  wmma_gemm_128<1><<<grid128(M_BT, D_MODEL), blk256, 0, stream>>>(
      ob, wprjb, b_proj, projf, M_BT, D_MODEL, D_MODEL);

  // y = x + scatter(proj)  -> d_out
  scatter_residual<<<(int)(((long)M_Y * D_MODEL + 255) / 256), 256, 0, stream>>>(
      x, projf, yOut);

  // MLP: h1 = LN(y); h2 = gelu(h1 @ W_fc1^T + b_fc1); y += h2 @ W_fc2^T + b_fc2
  ln_to_bf16<<<M_Y, blk256, 0, stream>>>(yOut, g2, be2, h1b);
  wmma_gemm_128<3><<<grid128(M_Y, D_HID), blk256, 0, stream>>>(
      h1b, wfc1b, b_fc1, h2b, M_Y, D_HID, D_MODEL);
  wmma_gemm_128<4><<<grid128(M_Y, D_MODEL), blk256, 0, stream>>>(
      h2b, wfc2b, b_fc2, yOut, M_Y, D_MODEL, D_HID);
}
